// Attn_69209103007798
// MI455X (gfx1250) — matrix operand order repaired and checked by reading, not compiled, not run on hardware
//
#include <hip/hip_runtime.h>
#include <math.h>
#include <stdint.h>

#define Hdim 1024
#define Bdim 32
#define Ldim 2048

typedef __attribute__((ext_vector_type(16))) __bf16 v16bf;
typedef __attribute__((ext_vector_type(8)))  __bf16 v8bf;
typedef __attribute__((ext_vector_type(8)))  float  v8f;

__device__ __forceinline__ float tanh_f(float x) {
#if __has_builtin(__builtin_amdgcn_tanhf)
    return __builtin_amdgcn_tanhf(x);
#else
    return tanhf(x);
#endif
}

#if __has_builtin(__builtin_amdgcn_global_load_async_to_lds_b128)
#define HAVE_ASYNC_LDS 1
typedef int v4i __attribute__((ext_vector_type(4)));
typedef __attribute__((address_space(1))) v4i* as1_v4i;   // global (__device__)
typedef __attribute__((address_space(3))) v4i* as3_v4i;   // LDS (__shared__)
__device__ __forceinline__ void async_b128(const void* g, void* l) {
    __builtin_amdgcn_global_load_async_to_lds_b128(
        (as1_v4i)(uintptr_t)g, (as3_v4i)(uintptr_t)l, 0, 0);
}
__device__ __forceinline__ void wait_async0() {
#if __has_builtin(__builtin_amdgcn_s_wait_asynccnt)
    __builtin_amdgcn_s_wait_asynccnt(0);
#else
    asm volatile("s_wait_asynccnt 0x0" ::: "memory");
#endif
}
#else
#define HAVE_ASYNC_LDS 0
__device__ __forceinline__ void wait_async0() {}
#endif

__device__ __forceinline__ v8bf cvt8(float4 a, float4 b) {
    v8bf r;
    r[0] = (__bf16)a.x; r[1] = (__bf16)a.y; r[2] = (__bf16)a.z; r[3] = (__bf16)a.w;
    r[4] = (__bf16)b.x; r[5] = (__bf16)b.y; r[6] = (__bf16)b.z; r[7] = (__bf16)b.w;
    return r;
}

__device__ __forceinline__ v16bf load_bfrag(const __bf16* p) {
    v16bf r;
    ((uint4*)&r)[0] = *(const uint4*)p;
    ((uint4*)&r)[1] = *(const uint4*)(p + 8);
    return r;
}

// ---------------------------------------------------------------------------
// Kernel 1: convert w_e = attn_w[:, H:2H] (f32) -> bf16 [o][h]; zero scores
// ---------------------------------------------------------------------------
__global__ __launch_bounds__(256) void prep_wbf_kernel(
    const float* __restrict__ attn_w, __bf16* __restrict__ wbf,
    float* __restrict__ scores) {
    int idx = blockIdx.x * 256 + threadIdx.x;          // 0 .. H*H-1
    int o = idx >> 10;
    int h = idx & 1023;
    wbf[idx] = (__bf16)attn_w[(size_t)o * (2 * Hdim) + Hdim + h];
    if (idx < Bdim * Ldim) scores[idx] = 0.f;          // atomically accumulated later
}

// ---------------------------------------------------------------------------
// Kernel 2: sbias[b][o] = hidden[b]·w_h[o] + attn_b[o]; also zero d_out
// ---------------------------------------------------------------------------
__global__ __launch_bounds__(256) void prep_sbias_kernel(
    const float* __restrict__ hidden, const float* __restrict__ attn_w,
    const float* __restrict__ attn_b, float* __restrict__ sbias,
    float* __restrict__ out) {
    int o = blockIdx.x * 256 + threadIdx.x;            // grid = 4 -> o in [0,1024)
    const float* wrow = attn_w + (size_t)o * (2 * Hdim);
    float bias = attn_b[o];
    for (int b = 0; b < Bdim; ++b) {
        const float* hb = hidden + (size_t)b * Hdim;
        float a0 = 0.f, a1 = 0.f, a2 = 0.f, a3 = 0.f;
        for (int h = 0; h < Hdim; h += 4) {
            float4 w4 = *(const float4*)(wrow + h);
            float4 h4 = *(const float4*)(hb + h);
            a0 += w4.x * h4.x; a1 += w4.y * h4.y;
            a2 += w4.z * h4.z; a3 += w4.w * h4.w;
        }
        sbias[(size_t)b * Hdim + o] = a0 + a1 + a2 + a3 + bias;
        out[(size_t)b * Hdim + o]   = 0.f;
    }
}

// ---------------------------------------------------------------------------
// Kernel 3: scores[b][l] += sum_{o in Nchunk} tanh(sbias[b][o] + enc_m·w_e[o])·v[o]
//   Block = 64 M-rows x 256 N-cols; 8 waves; wave = 2 N-tiles x 4 M-tiles.
//   Pipeline per 128-wide K-chunk:
//     - async (DMA) prefetch of next f32 A-chunk into rawA[1-cur]  (overlaps all)
//     - one f32->bf16 conversion pass rawA[cur] -> Abf              (once/element)
//     - 4 K-steps x 8 WMMA, A frags = 2x ds_load_b128, B pipelined 1 step ahead
// ---------------------------------------------------------------------------
__global__ __launch_bounds__(256) void attn_scores_kernel(
    const float*  __restrict__ enc,    // (L*B, H); row m = l*B + b
    const __bf16* __restrict__ wbf,    // (H, H)  [o][h]
    const float*  __restrict__ sbias,  // (B, H)
    const float*  __restrict__ vvec,   // (H)
    float*        __restrict__ scores) // (B, L), pre-zeroed, atomic +=
{
    __shared__ float  rawA[2][64][132];   // f32 double buffer (+4 pad)
    __shared__ __bf16 Abf[64][136];       // bf16 tile (+8 pad)
    __shared__ float  score_s[64];

    const int tid   = threadIdx.x;
    const int lane  = tid & 31;
    const int wave  = tid >> 5;
    const int lrow  = lane & 15;       // M (A) / N (B) index within 16x16 tile
    const int hh    = lane >> 4;       // K-half selector
    const int nblk  = blockIdx.x & 3;            // fastest: N chunk (L2 co-residency)
    const int mblk  = blockIdx.x >> 2;
    const int mbase = mblk * 64;
    const int nwave = nblk * 256 + wave * 32;    // wave owns 2 N-tiles

    const int srow = tid >> 2;                   // staging: 4 threads per row
    const int scol = (tid & 3) * 32;

    if (tid < 64) score_s[tid] = 0.f;

    v8f acc[2][4];
#pragma unroll
    for (int n = 0; n < 2; ++n)
#pragma unroll
        for (int mt = 0; mt < 4; ++mt)
#pragma unroll
            for (int r = 0; r < 8; ++r) acc[n][mt][r] = 0.f;

    const float* srcbase = enc + (size_t)(mbase + srow) * Hdim + scol;

    // prologue: stage chunk 0 into rawA[0]
    {
        const float* src = srcbase;
        float* dst = &rawA[0][srow][scol];
#if HAVE_ASYNC_LDS
#pragma unroll
        for (int j = 0; j < 8; ++j) async_b128(src + j * 4, dst + j * 4);
#else
#pragma unroll
        for (int j = 0; j < 8; ++j)
            *(float4*)(dst + j * 4) = *(const float4*)(src + j * 4);
#endif
    }

#pragma unroll 1
    for (int c = 0; c < Hdim / 128; ++c) {
        const int k0  = c * 128;
        const int cur = c & 1;

        wait_async0();
        __syncthreads();   // rawA[cur] ready; prev chunk's Abf reads done

        if (c + 1 < Hdim / 128) {   // DMA-prefetch next chunk (overlaps everything)
            const float* src = srcbase + k0 + 128;
            float* dst = &rawA[1 - cur][srow][scol];
#if HAVE_ASYNC_LDS
#pragma unroll
            for (int j = 0; j < 8; ++j) async_b128(src + j * 4, dst + j * 4);
#else
#pragma unroll
            for (int j = 0; j < 8; ++j)
                *(float4*)(dst + j * 4) = *(const float4*)(src + j * 4);
#endif
        }

        // hoisted B loads for ks=0 (independent of LDS conversion below)
        const __bf16* bp0 = wbf + (size_t)(nwave + lrow)      * Hdim + k0 + hh * 16;
        const __bf16* bp1 = wbf + (size_t)(nwave + 16 + lrow) * Hdim + k0 + hh * 16;
        v16bf b0 = load_bfrag(bp0);
        v16bf b1 = load_bfrag(bp1);

        // conversion pass: rawA[cur] -> Abf (once per element per block)
        {
            const float* s = &rawA[cur][srow][scol];
            __bf16* d = &Abf[srow][scol];
#pragma unroll
            for (int j = 0; j < 2; ++j) {
                float4 f0 = *(const float4*)(s + j * 16);
                float4 f1 = *(const float4*)(s + j * 16 + 4);
                float4 f2 = *(const float4*)(s + j * 16 + 8);
                float4 f3 = *(const float4*)(s + j * 16 + 12);
                *(v8bf*)(d + j * 16)     = cvt8(f0, f1);
                *(v8bf*)(d + j * 16 + 8) = cvt8(f2, f3);
            }
        }
        __syncthreads();   // Abf ready

#pragma unroll
        for (int ks = 0; ks < 4; ++ks) {
            v16bf c0 = b0, c1 = b1;
            if (ks < 3) {               // pipeline B one K-step ahead
                b0 = load_bfrag(bp0 + (ks + 1) * 32);
                b1 = load_bfrag(bp1 + (ks + 1) * 32);
            }
            // A fragments: elems 0..15 = K hh*16+0..15
            v16bf af[4];
#pragma unroll
            for (int mt = 0; mt < 4; ++mt) {
                const __bf16* ap = &Abf[mt * 16 + lrow][ks * 32 + hh * 16];
                ((uint4*)&af[mt])[0] = *(const uint4*)ap;
                ((uint4*)&af[mt])[1] = *(const uint4*)(ap + 8);
            }
#pragma unroll
            for (int mt = 0; mt < 4; ++mt)
                acc[0][mt] = __builtin_amdgcn_wmma_f32_16x16x32_bf16(
                    false, af[mt], false, c0, (short)0, acc[0][mt], false, false);
#pragma unroll
            for (int mt = 0; mt < 4; ++mt)
                acc[1][mt] = __builtin_amdgcn_wmma_f32_16x16x32_bf16(
                    false, af[mt], false, c1, (short)0, acc[1][mt], false, false);
        }
    }

    // Epilogue: + sbias, tanh, * v, reduce over the block's N range.
    // C/D layout: VGPR r, lane t -> Mloc = mt*16 + r + 8*(t/16), N = nwave+n*16+(t&15)
    float p[4][8];
#pragma unroll
    for (int mt = 0; mt < 4; ++mt)
#pragma unroll
        for (int r = 0; r < 8; ++r) p[mt][r] = 0.f;

#pragma unroll
    for (int n = 0; n < 2; ++n) {
        int o = nwave + n * 16 + lrow;
        float vv = vvec[o];
        const float* sb = sbias + o;
#pragma unroll
        for (int mt = 0; mt < 4; ++mt)
#pragma unroll
            for (int r = 0; r < 8; ++r) {
                int b = (mt & 1) * 16 + r + 8 * hh;      // (mbase+Mloc) & 31
                float e = tanh_f(acc[n][mt][r] + sb[(size_t)b * Hdim]);
                p[mt][r] += e * vv;
            }
    }
    __syncthreads();
#pragma unroll
    for (int mt = 0; mt < 4; ++mt)
#pragma unroll
        for (int r = 0; r < 8; ++r)
            atomicAdd(&score_s[mt * 16 + r + 8 * hh], p[mt][r]);  // ds_add_f32
    __syncthreads();
    if (tid < 64) {
        int m = mbase + tid;                 // row = l*B + b
        atomicAdd(&scores[(size_t)(m & 31) * Ldim + (m >> 5)], score_s[tid]);
    }
}

// ---------------------------------------------------------------------------
// Kernel 4: in-place softmax over l for each b (32 blocks x 256 threads)
// ---------------------------------------------------------------------------
__global__ __launch_bounds__(256) void softmax_kernel(float* __restrict__ scores) {
    __shared__ float red[256];
    float* row = scores + (size_t)blockIdx.x * Ldim;
    int tid = threadIdx.x;

    float m = -INFINITY;
    for (int i = tid; i < Ldim; i += 256) m = fmaxf(m, row[i]);
    red[tid] = m; __syncthreads();
    for (int s = 128; s > 0; s >>= 1) {
        if (tid < s) red[tid] = fmaxf(red[tid], red[tid + s]);
        __syncthreads();
    }
    m = red[0]; __syncthreads();

    float vals[8]; float sum = 0.f;
#pragma unroll
    for (int j = 0; j < 8; ++j) {
        vals[j] = __expf(row[tid + j * 256] - m);
        sum += vals[j];
    }
    red[tid] = sum; __syncthreads();
    for (int s = 128; s > 0; s >>= 1) {
        if (tid < s) red[tid] += red[tid + s];
        __syncthreads();
    }
    float inv = 1.f / red[0];
#pragma unroll
    for (int j = 0; j < 8; ++j) row[tid + j * 256] = vals[j] * inv;
}

// ---------------------------------------------------------------------------
// Kernel 5: context[b][h] = sum_l attn[b][l] * enc[l][b][h]
//   L split 8-way across blocks; partials combined with global_atomic_add_f32.
// ---------------------------------------------------------------------------
__global__ __launch_bounds__(256) void context_kernel(
    const float* __restrict__ enc, const float* __restrict__ attn,
    float* __restrict__ out) {
    int b     = blockIdx.x & 31;
    int hblk  = (blockIdx.x >> 5) & 3;
    int chunk = blockIdx.x >> 7;            // 0..7
    int h     = hblk * 256 + threadIdx.x;
    int l0    = chunk * 256;

    const float* base = enc + (size_t)b * Hdim + h;
    const float* arow = attn + (size_t)b * Ldim;
    float acc = 0.f;
    for (int l = l0; l < l0 + 256; ++l)
        acc += arow[l] * base[(size_t)l * (Bdim * Hdim)];
    atomicAdd(out + (size_t)b * Hdim + h, acc);
}

// ---------------------------------------------------------------------------
extern "C" void kernel_launch(void* const* d_in, const int* in_sizes, int n_in,
                              void* d_out, int out_size, void* d_ws, size_t ws_size,
                              hipStream_t stream) {
    const float* hidden = (const float*)d_in[0];   // (B, H)
    const float* enc    = (const float*)d_in[1];   // (L, B, H) == (L*B, H)
    const float* attn_w = (const float*)d_in[2];   // (H, 2H)
    const float* attn_b = (const float*)d_in[3];   // (H)
    const float* vvec   = (const float*)d_in[4];   // (H)
    float* out = (float*)d_out;                    // (1, B, H)

    char* ws = (char*)d_ws;
    __bf16* wbf   = (__bf16*)ws;                               // 2 MB
    float*  sbias = (float*)(ws + (size_t)2 * 1024 * 1024);    // 128 KB
    float*  scores= (float*)(ws + (size_t)2 * 1024 * 1024 + 128 * 1024); // 256 KB

    prep_wbf_kernel  <<<(Hdim * Hdim) / 256, 256, 0, stream>>>(attn_w, wbf, scores);
    prep_sbias_kernel<<<Hdim / 256,          256, 0, stream>>>(hidden, attn_w,
                                                               attn_b, sbias, out);
    // grid: (M/64) * (N/256); N-chunk fastest for L2 co-residency of A stream
    attn_scores_kernel<<<((Ldim * Bdim) / 64) * 4, 256, 0, stream>>>(
        enc, wbf, sbias, vvec, scores);
    softmax_kernel   <<<Bdim,                256, 0, stream>>>(scores);
    context_kernel   <<<Bdim * (Hdim / 256) * 8, 256, 0, stream>>>(enc, scores, out);
}
